// BandSplitEncoder_69475390980338
// MI455X (gfx1250) — matrix operand order repaired and checked by reading, not compiled, not run on hardware
//
#include <hip/hip_runtime.h>

typedef __attribute__((ext_vector_type(16))) __bf16 v16bf;
typedef __attribute__((ext_vector_type(8)))  float  v8f;

union Frag16 { unsigned int u[8]; v16bf v; };

#define T_LEN 4096

// Band geometry (compile-time constants of the problem)
__constant__ int c_s[8]   = {0,8,16,24,32,48,64,96};
__constant__ int c_w[8]   = {8,8,8,8,16,16,32,161};
__constant__ int c_ss0[8] = {0,4,8,12,16,24,32,48};
__constant__ int c_w0[8]  = {4,4,4,4,8,8,16,80};
__constant__ int c_ss1[8] = {0,2,4,6,8,12,16,24};
__constant__ int c_w1[8]  = {2,2,2,2,4,4,8,40};
__constant__ int c_kpm[8] = {32,32,32,32,32,32,64,352}; // 2*w padded to 32
__constant__ int c_kpa[8] = {32,32,32,32,32,32,64,256}; // 2*w0+2*w1 padded to 32
__constant__ int c_ktm[8] = {0,1,2,3,4,5,6,8};          // main k-tile offsets (total 19)
__constant__ int c_kta[8] = {0,1,2,3,4,5,6,8};          // aux  k-tile offsets (total 16)

__device__ __forceinline__ unsigned short f2bf(float f) {
  unsigned int x = __float_as_uint(f);
  x += 0x7fffu + ((x >> 16) & 1u);          // round to nearest even
  return (unsigned short)(x >> 16);
}
__device__ __forceinline__ unsigned int pack_bf(float lo, float hi) {
  return (unsigned int)f2bf(lo) | ((unsigned int)f2bf(hi) << 16);
}

struct WPtrs {
  const float* bandW[8];
  const float* aux0W[8];
  const float* aux1W[8];
};

// ---------------------------------------------------------------------------
// Weight prep: f32 ragged (D,K) -> bf16 WMMA B-fragment layout, K padded to 32.
// Fragment dword index: ((ktile_global*16 + ntile)*32 + lane)*8 + j
// B order used here (contiguous k, as A): VGPR j, lanes 0-15: N=lane, K=2j,2j+1; lanes16-31: K+16.
// ---------------------------------------------------------------------------
__global__ void prep_weights_kernel(WPtrs wp,
                                    unsigned int* __restrict__ mainW,
                                    unsigned int* __restrict__ auxW) {
  const int MAIN_D = 19 * 4096;
  const int AUX_D  = 16 * 4096;
  int t = blockIdx.x * blockDim.x + threadIdx.x;
  if (t >= MAIN_D + AUX_D) return;
  bool isMain = (t < MAIN_D);
  int tt   = isMain ? t : (t - MAIN_D);
  int j    = tt & 7;
  int lane = (tt >> 3) & 31;
  int nt   = (tt >> 8) & 15;
  int ktg  = tt >> 12;
  int band = 0;
  if (isMain) { while (band < 7 && ktg >= c_ktm[band + 1]) band++; }
  else        { while (band < 7 && ktg >= c_kta[band + 1]) band++; }
  int kt = ktg - (isMain ? c_ktm[band] : c_kta[band]);
  int n  = nt * 16 + (lane & 15);
  int kp = kt * 32 + 2 * j + ((lane & 16) ? 16 : 0);
  float v0 = 0.f, v1 = 0.f;
  if (isMain) {
    int K = 2 * c_w[band];
    const float* W = wp.bandW[band];
    if (kp     < K) v0 = W[n * K + kp];
    if (kp + 1 < K) v1 = W[n * K + kp + 1];
    mainW[tt] = pack_bf(v0, v1);
  } else {
    int K0 = 2 * c_w0[band], K1 = 2 * c_w1[band];
    const float* W0 = wp.aux0W[band];
    const float* W1 = wp.aux1W[band];
    int k2 = kp;
    if (k2 < K0)               v0 = W0[n * K0 + k2];
    else if (k2 - K0 < K1)     v0 = W1[n * K1 + (k2 - K0)];
    k2 = kp + 1;
    if (k2 < K0)               v1 = W0[n * K0 + k2];
    else if (k2 - K0 < K1)     v1 = W1[n * K1 + (k2 - K0)];
    auxW[tt] = pack_bf(v0, v1);
  }
}

// ---------------------------------------------------------------------------
// Main fused kernel: one (band, 32-t-row) tile per block; 8 waves, each wave
// owns a 16x64 output sub-tile. bf16 WMMA GEMMs + layernorm + softmax mix.
// ---------------------------------------------------------------------------
#define SAM 181   // main A-tile row stride in dwords (odd -> bank-conflict-free)
#define SAA 129   // aux  A-tile row stride in dwords

__global__ __launch_bounds__(256) void bandsplit_wmma_kernel(
    const float* __restrict__ pr,  const float* __restrict__ pi,
    const float* __restrict__ a1r, const float* __restrict__ a1i,
    const float* __restrict__ a2r, const float* __restrict__ a2i,
    const float* __restrict__ band_b, const float* __restrict__ gamma_,
    const float* __restrict__ beta_,  const float* __restrict__ aux_b,
    const float* __restrict__ scale_w,
    const unsigned int* __restrict__ mainW,
    const unsigned int* __restrict__ auxW,
    float* __restrict__ out)
{
  __shared__ unsigned int sAm[32 * SAM];
  __shared__ unsigned int sAa[32 * SAA];
  __shared__ float sStats[64];

  const int band = blockIdx.y;
  const int b    = blockIdx.z;
  const int t0   = blockIdx.x * 32;
  const int tid  = threadIdx.x;

  // softmax(scale_weights) — 3 elements, computed by every thread
  float s0 = scale_w[0], s1 = scale_w[1], s2 = scale_w[2];
  float mx = fmaxf(s0, fmaxf(s1, s2));
  float e0 = __expf(s0 - mx), e1 = __expf(s1 - mx), e2 = __expf(s2 - mx);
  float ei = 1.f / (e0 + e1 + e2);
  float sw0 = e0 * ei, sw1 = e1 * ei, sw2 = e2 * ei;

  const int w   = c_w[band],  s   = c_s[band],   Kpm = c_kpm[band];
  const int w0  = c_w0[band], ss0 = c_ss0[band];
  const int w1  = c_w1[band], ss1 = c_ss1[band], Kpa = c_kpa[band];

  // ---- stage main A-tile: [32 rows][Kpm] bf16, rows = t, k = [real|imag|pad]
  unsigned short* Am16 = (unsigned short*)sAm;
  for (int idx = tid; idx < 32 * Kpm; idx += 256) {
    int row = idx / Kpm;
    int k   = idx - row * Kpm;
    float v = 0.f;
    int base = (b * T_LEN + t0 + row) * 257 + s;
    if (k < w)            v = pr[base + k];
    else if (k < 2 * w)   v = pi[base + (k - w)];
    Am16[row * (2 * SAM) + k] = f2bf(v);
  }
  // ---- stage aux A-tile: k = [re0|im0|re1|im1|pad], sw folded in; aux mem is
  // [B,F,T] so iterate k-major / t-minor for coalesced global reads.
  unsigned short* Aa16 = (unsigned short*)sAa;
  for (int idx = tid; idx < 32 * Kpa; idx += 256) {
    int kk  = idx >> 5;
    int row = idx & 31;
    int t   = t0 + row;
    float v = 0.f;
    if (kk < w0)                 v = sw1 * a1r[(b * 129 + ss0 + kk) * T_LEN + t];
    else if (kk < 2 * w0)        v = sw1 * a1i[(b * 129 + ss0 + (kk - w0)) * T_LEN + t];
    else if (kk < 2 * w0 + w1)   v = sw2 * a2r[(b * 65 + ss1 + (kk - 2 * w0)) * T_LEN + t];
    else if (kk < 2 * w0 + 2*w1) v = sw2 * a2i[(b * 65 + ss1 + (kk - 2 * w0 - w1)) * T_LEN + t];
    Aa16[row * (2 * SAA) + kk] = f2bf(v);
  }
  if (tid < 64) sStats[tid] = 0.f;
  __syncthreads();

  const int lane  = tid & 31;
  const int wv    = tid >> 5;
  const int mbase = (wv >> 2) * 16;   // waves 0-3: rows 0-15, 4-7: rows 16-31
  const int nq    = wv & 3;           // 64-column quarter of D
  const int m     = mbase + (lane & 15);
  const int up    = lane >> 4;

  v8f z8 = {0.f,0.f,0.f,0.f,0.f,0.f,0.f,0.f};
  v8f accM[4] = {z8, z8, z8, z8};
  v8f accA[4] = {z8, z8, z8, z8};

  // A order used here (contiguous k, as B): VGPR j: K dword j + 8*up
  // ---- main GEMM ----
  {
    const int nkt = Kpm >> 5;
    const int kb  = c_ktm[band];
    for (int kt = 0; kt < nkt; ++kt) {
      Frag16 af;
      #pragma unroll
      for (int j = 0; j < 8; ++j)
        af.u[j] = sAm[m * SAM + kt * 16 + j + 8 * up];
      #pragma unroll
      for (int nt = 0; nt < 4; ++nt) {
        const uint4* bp = (const uint4*)(mainW +
            (size_t)(((kb + kt) * 16 + (nq * 4 + nt)) * 32 + lane) * 8);
        uint4 q0 = bp[0], q1 = bp[1];
        Frag16 bfr;
        bfr.u[0]=q0.x; bfr.u[1]=q0.y; bfr.u[2]=q0.z; bfr.u[3]=q0.w;
        bfr.u[4]=q1.x; bfr.u[5]=q1.y; bfr.u[6]=q1.z; bfr.u[7]=q1.w;
        accM[nt] = __builtin_amdgcn_wmma_f32_16x16x32_bf16(
            false, af.v, false, bfr.v, (short)0, accM[nt], false, false);
      }
    }
  }
  // ---- fused aux GEMM (both scales, pre-scaled A) ----
  {
    const int nkt = Kpa >> 5;
    const int kb  = c_kta[band];
    for (int kt = 0; kt < nkt; ++kt) {
      Frag16 af;
      #pragma unroll
      for (int j = 0; j < 8; ++j)
        af.u[j] = sAa[m * SAA + kt * 16 + j + 8 * up];
      #pragma unroll
      for (int nt = 0; nt < 4; ++nt) {
        const uint4* bp = (const uint4*)(auxW +
            (size_t)(((kb + kt) * 16 + (nq * 4 + nt)) * 32 + lane) * 8);
        uint4 q0 = bp[0], q1 = bp[1];
        Frag16 bfr;
        bfr.u[0]=q0.x; bfr.u[1]=q0.y; bfr.u[2]=q0.z; bfr.u[3]=q0.w;
        bfr.u[4]=q1.x; bfr.u[5]=q1.y; bfr.u[6]=q1.z; bfr.u[7]=q1.w;
        accA[nt] = __builtin_amdgcn_wmma_f32_16x16x32_bf16(
            false, af.v, false, bfr.v, (short)0, accA[nt], false, false);
      }
    }
  }

  // ---- epilogue: bias, layernorm stats (row sums across D=256), mix, store
  const int nc = lane & 15;
  float bb[4], gm[4], bt[4], ab[4];
  #pragma unroll
  for (int nt = 0; nt < 4; ++nt) {
    int n = nq * 64 + nt * 16 + nc;
    bb[nt] = band_b[band * 256 + n];
    gm[nt] = gamma_[band * 256 + n];
    bt[nt] = beta_ [band * 256 + n];
    ab[nt] = sw1 * aux_b[band * 256 + n] + sw2 * aux_b[2048 + band * 256 + n];
  }
  #pragma unroll
  for (int v = 0; v < 8; ++v) {
    float p = 0.f, q = 0.f;
    #pragma unroll
    for (int nt = 0; nt < 4; ++nt) {
      float x = accM[nt][v] + bb[nt];
      accM[nt][v] = x;
      p += x; q += x * x;
    }
    #pragma unroll
    for (int msk = 1; msk < 16; msk <<= 1) {     // 16-lane butterfly per half
      p += __shfl_xor(p, msk, 32);
      q += __shfl_xor(q, msk, 32);
    }
    if ((lane & 15) == v) {                      // one lane per (half, row)
      int row = mbase + 8 * up + v;
      atomicAdd(&sStats[row], p);                // ds_add_f32 across N-waves
      atomicAdd(&sStats[32 + row], q);
    }
  }
  __syncthreads();
  #pragma unroll
  for (int v = 0; v < 8; ++v) {
    int row   = mbase + 8 * up + v;
    float mean = sStats[row] * (1.f / 256.f);
    float var  = sStats[32 + row] * (1.f / 256.f) - mean * mean;
    float rinv = rsqrtf(var + 1e-5f);
    #pragma unroll
    for (int nt = 0; nt < 4; ++nt) {
      int n = nq * 64 + nt * 16 + nc;
      float hn = (accM[nt][v] - mean) * rinv * gm[nt] + bt[nt];
      float o  = hn * sw0 + accA[nt][v] + ab[nt];
      size_t addr = ((size_t)((b * T_LEN + t0 + row) * 8 + band)) * 256 + n;
      out[addr] = o;
    }
  }
}

extern "C" void kernel_launch(void* const* d_in, const int* in_sizes, int n_in,
                              void* d_out, int out_size, void* d_ws, size_t ws_size,
                              hipStream_t stream) {
  (void)in_sizes; (void)n_in; (void)out_size; (void)ws_size;
  // workspace: 19*4096 + 16*4096 dwords of bf16-packed WMMA B fragments
  unsigned int* mainW = (unsigned int*)d_ws;
  unsigned int* auxW  = mainW + 19 * 4096;

  WPtrs wp;
  for (int i = 0; i < 8; ++i) wp.bandW[i] = (const float*)d_in[6 + i];
  for (int i = 0; i < 8; ++i) wp.aux0W[i] = (const float*)d_in[17 + i];
  for (int i = 0; i < 8; ++i) wp.aux1W[i] = (const float*)d_in[25 + i];

  prep_weights_kernel<<<560, 256, 0, stream>>>(wp, mainW, auxW);

  dim3 grid(T_LEN / 32, 8, 8);  // (t-tiles, bands, batch)
  bandsplit_wmma_kernel<<<grid, 256, 0, stream>>>(
      (const float*)d_in[0], (const float*)d_in[1],
      (const float*)d_in[2], (const float*)d_in[3],
      (const float*)d_in[4], (const float*)d_in[5],
      (const float*)d_in[14], (const float*)d_in[15], (const float*)d_in[16],
      (const float*)d_in[33], (const float*)d_in[34],
      mainW, auxW, (float*)d_out);
}
